// SAGE_46574625358105
// MI455X (gfx1250) — matrix operand order repaired and checked by reading, not compiled, not run on hardware
//
#include <hip/hip_runtime.h>
#include <hip/hip_fp16.h>

typedef __attribute__((ext_vector_type(16))) _Float16 v16h;
typedef __attribute__((ext_vector_type(8)))  float    v8f;

__device__ __forceinline__ void atomAddF(float* p, float v) {
    // lowers to global_atomic_add_f32 (no return -> STOREcnt) on gfx1250
    unsafeAtomicAdd(p, v);
}

// ---------------------------------------------------------------- degree
__global__ void k_deg_init(float* __restrict__ deg, int N) {
    int i = blockIdx.x * blockDim.x + threadIdx.x;
    if (i < N) deg[i] = 1.0f;               // self-loop contributes 1
}

__global__ void k_deg_accum(const int* __restrict__ col, float* __restrict__ deg, int E) {
    int e = blockIdx.x * blockDim.x + threadIdx.x;
    if (e < E) atomAddF(&deg[col[e]], 1.0f);
}

__global__ void k_dinv(float* __restrict__ deg, int N) {
    int i = blockIdx.x * blockDim.x + threadIdx.x;
    if (i < N) deg[i] = __frsqrt_rn(deg[i]);   // deg >= 1 always
}

// ---------------------------------------------------------------- B prepack
// Convert W1 / Wf1 (each [128 x 64] f32 row-major) into f16 WMMA B-fragments,
// laid out exactly as each lane consumes them:
//   Bp[((mat*4 + ntile)*4 + kstep)*32 + lane][j] , j = 0..15 (contiguous 32 B)
//   fragment element j: k = kstep*32 + (lane/16)*16 + j , n = ntile*16 + lane%16
// Total: 2 * 4 * 4 * 32 * 16 f16 = 32 KB (stays in L2, read by all blocks).
__global__ void k_prepackB(const float* __restrict__ W1, const float* __restrict__ Wf1,
                           _Float16* __restrict__ Bp) {
    int idx = blockIdx.x * blockDim.x + threadIdx.x;      // 0..1023
    int mat   = idx >> 9;
    int ntile = (idx >> 7) & 3;
    int ks    = (idx >> 5) & 3;
    int lane  = idx & 31;
    const float* W = mat ? Wf1 : W1;
    int kBase = ks * 32 + (lane >> 4) * 16;
    int n     = ntile * 16 + (lane & 15);
    _Float16* dst = Bp + (size_t)idx * 16;
#pragma unroll
    for (int j = 0; j < 16; ++j)
        dst[j] = (_Float16)W[(size_t)(kBase + j) * 64 + n];
}

// ---------------------------------------------------------------- fused WMMA GEMM
// One pass over x computes BOTH products:
//   XW1[N x 64]      = x @ W1                  (-> scatter input)
//   H[:, 64:128]     = x @ Wf1 + bf1           (-> fc branch)
// Block = 4 waves; wave w owns N-tile w; blockIdx.x = M-tile (16 rows).
// Each K-step: one A fragment feeds two v_wmma_f32_16x16x32_f16.
__global__ void k_gemm_dual(const float* __restrict__ X, const _Float16* __restrict__ Bp,
                            const float* __restrict__ bf1,
                            float* __restrict__ XW1, float* __restrict__ H, int N) {
    const int lane  = threadIdx.x & 31;
    const int wave  = threadIdx.x >> 5;
    const int half  = lane >> 4;          // 0: lanes 0-15, 1: lanes 16-31
    const int l16   = lane & 15;
    const int mBase = blockIdx.x * 16;

    int aRow = mBase + l16;
    if (aRow >= N) aRow = N - 1;          // clamp for tail tile (stores masked)
    const float* __restrict__ xrow = X + (size_t)aRow * 128;

    v8f acc0 = {};   // x @ W1
    v8f acc1 = {};   // x @ Wf1
#pragma unroll
    for (int ks = 0; ks < 4; ++ks) {
        const int k0 = ks * 32;
        // A fragment (16x32 f16): lane l, elem j -> k = k0 + half*16 + j
        v16h a;
        const float* xa = xrow + k0 + half * 16;
#pragma unroll
        for (int j = 0; j < 8; ++j) {
            a[j]     = (_Float16)xa[j];
            a[j + 8] = (_Float16)xa[8 + j];
        }
        // Prepacked B fragments: one contiguous 32-byte load each
        const v16h b0 = *(const v16h*)(Bp + (size_t)((((0 * 4 + wave) * 4 + ks) * 32) + lane) * 16);
        const v16h b1 = *(const v16h*)(Bp + (size_t)((((1 * 4 + wave) * 4 + ks) * 32) + lane) * 16);

        acc0 = __builtin_amdgcn_wmma_f32_16x16x32_f16(false, a, false, b0, (short)0, acc0, false, false);
        acc1 = __builtin_amdgcn_wmma_f32_16x16x32_f16(false, a, false, b1, (short)0, acc1, false, false);
    }
    // D layout: lane l, elem r -> m = mBase + half*8 + r, n = l16
    const int n  = wave * 16 + l16;
    const float bv = bf1[n];
#pragma unroll
    for (int r = 0; r < 8; ++r) {
        int m = mBase + half * 8 + r;
        if (m < N) {
            XW1[(size_t)m * 64 + n]       = acc0[r];
            H[(size_t)m * 128 + 64 + n]   = acc1[r] + bv;
        }
    }
}

// ---------------------------------------------------------------- self-loop init
// H[i, c] = b1[c] + dinv[i]^2 * xw1[i, c]   for c in [0,64)
__global__ void k_self_init(const float* __restrict__ xw1, const float* __restrict__ dinv,
                            const float* __restrict__ b1, float* __restrict__ H, int N) {
    int t = blockIdx.x * blockDim.x + threadIdx.x;
    if (t < N * 64) {
        int i = t >> 6, c = t & 63;
        float d = dinv[i];
        H[(size_t)i * 128 + c] = b1[c] + d * d * xw1[t];
    }
}

// ---------------------------------------------------------------- feature scatter
// One wave per edge: H[col, 0:64] += norm * xw1[row, 0:64]   (2 floats / lane)
__global__ void k_scatter_feat(const int* __restrict__ row, const int* __restrict__ col,
                               const float* __restrict__ dinv, const float* __restrict__ XW,
                               float* __restrict__ H, int E) {
    int gw     = (int)((blockIdx.x * blockDim.x + threadIdx.x) >> 5);
    int lane   = threadIdx.x & 31;
    int nwaves = (int)((gridDim.x * blockDim.x) >> 5);
    for (int e = gw; e < E; e += nwaves) {
        int r = row[e], c = col[e];
        float w = dinv[r] * dinv[c];
        const float2 v = ((const float2*)(XW + (size_t)r * 64))[lane];
        float* dst = H + (size_t)c * 128 + lane * 2;
        atomAddF(dst,     v.x * w);
        atomAddF(dst + 1, v.y * w);
    }
}

// ---------------------------------------------------------------- node stage 2
// One wave per node: ReLU on load, z = h.W2, f = h.Wf2; out[i] = f+bf2+b2+dinv^2*z
__global__ void k_node2(const float* __restrict__ H, const float* __restrict__ dinv,
                        const float* __restrict__ W2, const float* __restrict__ Wf2,
                        const float* __restrict__ b2, const float* __restrict__ bf2,
                        float* __restrict__ Z, float* __restrict__ out, int N) {
    int gw     = (int)((blockIdx.x * blockDim.x + threadIdx.x) >> 5);
    int lane   = threadIdx.x & 31;
    int nwaves = (int)((gridDim.x * blockDim.x) >> 5);
    for (int i = gw; i < N; i += nwaves) {
        const float* hrow = H + (size_t)i * 128;
        float h0 = fmaxf(hrow[lane],      0.f);
        float h1 = fmaxf(hrow[lane + 32], 0.f);
        float h2 = fmaxf(hrow[lane + 64], 0.f);
        float h3 = fmaxf(hrow[lane + 96], 0.f);
        float z = h0 * W2[lane]  + h1 * W2[lane + 32]  + h2 * W2[lane + 64]  + h3 * W2[lane + 96];
        float f = h0 * Wf2[lane] + h1 * Wf2[lane + 32] + h2 * Wf2[lane + 64] + h3 * Wf2[lane + 96];
#pragma unroll
        for (int off = 16; off > 0; off >>= 1) {
            z += __shfl_xor(z, off, 32);
            f += __shfl_xor(f, off, 32);
        }
        if (lane == 0) {
            float d = dinv[i];
            Z[i]   = z;
            out[i] = f + bf2[0] + b2[0] + d * d * z;
        }
    }
}

// ---------------------------------------------------------------- scalar scatter
__global__ void k_scatter_scalar(const int* __restrict__ row, const int* __restrict__ col,
                                 const float* __restrict__ dinv, const float* __restrict__ Z,
                                 float* __restrict__ out, int E) {
    int e = blockIdx.x * blockDim.x + threadIdx.x;
    if (e < E) {
        int r = row[e], c = col[e];
        atomAddF(&out[c], dinv[r] * dinv[c] * Z[r]);
    }
}

extern "C" void kernel_launch(void* const* d_in, const int* in_sizes, int n_in,
                              void* d_out, int out_size, void* d_ws, size_t ws_size,
                              hipStream_t stream) {
    const float* x   = (const float*)d_in[0];
    const int*   ei  = (const int*)d_in[1];     // [2, E] (int32; JAX x64 disabled)
    const float* W1  = (const float*)d_in[2];
    const float* b1  = (const float*)d_in[3];
    const float* Wf1 = (const float*)d_in[4];
    const float* bf1 = (const float*)d_in[5];
    const float* W2  = (const float*)d_in[6];
    const float* b2  = (const float*)d_in[7];
    const float* Wf2 = (const float*)d_in[8];
    const float* bf2 = (const float*)d_in[9];
    float* out = (float*)d_out;

    const int N = in_sizes[0] / 128;            // 100000
    const int E = in_sizes[1] / 2;              // 1600000
    const int* row = ei;                        // sources
    const int* col = ei + E;                    // targets

    // workspace layout (floats): dinv[N] | z[N] | xw1[N*64] | h[N*128] | Bp(32KB)
    float* ws   = (float*)d_ws;
    float* dinv = ws;
    float* z    = ws + (size_t)N;
    float* xw1  = ws + (size_t)2 * N;
    float* h    = ws + (size_t)2 * N + (size_t)64 * N;
    _Float16* Bp = (_Float16*)(ws + (size_t)194 * N);

    const int TB = 256;
    // 1. degrees -> dinv   (runs concurrently-safe w.r.t. prepack: different buffers,
    //    stream order still serializes; fine)
    k_deg_init <<<(N + TB - 1) / TB, TB, 0, stream>>>(dinv, N);
    k_deg_accum<<<(E + TB - 1) / TB, TB, 0, stream>>>(col, dinv, E);
    k_dinv     <<<(N + TB - 1) / TB, TB, 0, stream>>>(dinv, N);

    // 2. prepack W1/Wf1 into f16 B-fragments, then fused dual GEMM over x
    k_prepackB<<<4, 256, 0, stream>>>(W1, Wf1, Bp);
    int mTiles = (N + 15) / 16;
    k_gemm_dual<<<mTiles, 128, 0, stream>>>(x, Bp, bf1, xw1, h, N);

    // 3. h[:,:64] = b1 + dinv^2 * xw1  (self-loop), then edge scatter-add
    k_self_init<<<((N * 64) + TB - 1) / TB, TB, 0, stream>>>(xw1, dinv, b1, h, N);
    k_scatter_feat<<<(E + 7) / 8, TB, 0, stream>>>(row, col, dinv, xw1, h, E);

    // 4. per-node dots (ReLU fused on load) -> z, out(node-local part)
    k_node2<<<(N + 7) / 8, TB, 0, stream>>>(h, dinv, W2, Wf2, b2, bf2, z, out, N);

    // 5. out[col] += norm * z[row]
    k_scatter_scalar<<<(E + TB - 1) / TB, TB, 0, stream>>>(row, col, dinv, z, out, E);
}
